// FasterRCNN_4372276708011
// MI455X (gfx1250) — matrix operand order repaired and checked by reading, not compiled, not run on hardware
//
#include <hip/hip_runtime.h>
#include <hip/hip_bf16.h>
#include <math.h>

// ---------------- constants ----------------
#define NLEV   5
#define BATCH  2
#define PRE_NMS  1000
#define POST_NMS 1000
#define TOTA   279279          // sum of per-level anchor counts
#define CAND_MAX 8192
#define SORT_N   8192
#define CCH    256
#define DIMX   12544           // 7*7*256
#define MLP    1024
#define NCLS   91
#define MROWS  2048            // padded GEMM M
#define MVALID 2000            // B * POST_NMS
#define NMS_TH 0.7f
#define BBOX_CLIP 4.1351665f   // log(1000/16)

__constant__ int   c_H[NLEV]      = {208,104,52,26,13};
__constant__ int   c_W[NLEV]      = {336,168,84,42,21};
__constant__ int   c_N[NLEV]      = {209664,52416,13104,3276,819};
__constant__ int   c_nl[NLEV]     = {1000,1000,1000,1000,819};
__constant__ float c_stride[NLEV] = {4.f,8.f,16.f,32.f,64.f};

// ---------------- helpers ----------------
__device__ __forceinline__ unsigned short f2bf(float f) {
  unsigned u = __float_as_uint(f);
  unsigned r = u + 0x7fffu + ((u >> 16) & 1u);
  return (unsigned short)(r >> 16);
}
__device__ __forceinline__ float bf2f(unsigned short s) {
  return __uint_as_float(((unsigned)s) << 16);
}
__device__ __forceinline__ unsigned sortkey(float f) {
  unsigned u = __float_as_uint(f);
  return (u & 0x80000000u) ? ~u : (u | 0x80000000u);
}
__device__ __forceinline__ float unsortkey(unsigned s) {
  unsigned u = (s & 0x80000000u) ? (s & 0x7fffffffu) : ~s;
  return __uint_as_float(u);
}

typedef __attribute__((ext_vector_type(16))) __bf16 v16bf;
typedef __attribute__((ext_vector_type(8)))  float  v8f;

// ---------------- stage 1: scores ----------------
__global__ void k_sigmoid(const float* __restrict__ score, float* __restrict__ s_flat,
                          int off, int N) {
  int gid = blockIdx.x * blockDim.x + threadIdx.x;
  if (gid >= BATCH * N) return;
  int b = gid / N, g = gid % N;
  float v = score[gid];
  s_flat[(size_t)b * TOTA + off + g] = 1.0f / (1.0f + expf(-v));
}

__global__ void k_hist(const float* __restrict__ s_flat, int* __restrict__ hist,
                       int off, int N, int li) {
  int gid = blockIdx.x * blockDim.x + threadIdx.x;
  if (gid >= BATCH * N) return;
  int b = gid / N, g = gid % N;
  float s = s_flat[(size_t)b * TOTA + off + g];
  int bin = (int)(s * 1024.0f);
  bin = bin < 0 ? 0 : (bin > 1023 ? 1023 : bin);
  atomicAdd(&hist[(b * NLEV + li) * 1024 + bin], 1);
}

__global__ void k_thresh(const int* __restrict__ hist, int* __restrict__ thr) {
  int bl = blockIdx.x;            // b*5 + li
  int li = bl % NLEV;
  if (threadIdx.x == 0) {
    int need = c_nl[li], cum = 0, t = 0;
    for (int bin = 1023; bin >= 0; --bin) {
      cum += hist[bl * 1024 + bin];
      if (cum >= need) { t = bin; break; }
    }
    thr[bl] = t;
  }
}

__global__ void k_compact(const float* __restrict__ s_flat, const int* __restrict__ thr,
                          int* __restrict__ cnt, float* __restrict__ cval,
                          int* __restrict__ cidx, int off, int N, int li) {
  int gid = blockIdx.x * blockDim.x + threadIdx.x;
  if (gid >= BATCH * N) return;
  int b = gid / N, g = gid % N;
  int bl = b * NLEV + li;
  float s = s_flat[(size_t)b * TOTA + off + g];
  int bin = (int)(s * 1024.0f);
  bin = bin < 0 ? 0 : (bin > 1023 ? 1023 : bin);
  if (bin >= thr[bl]) {
    int pos = atomicAdd(&cnt[bl], 1);
    if (pos < CAND_MAX) {
      cval[(size_t)bl * CAND_MAX + pos] = s;
      cidx[(size_t)bl * CAND_MAX + pos] = g;
    }
  }
}

// ---------------- bitonic sort (descending, 8192 keys) ----------------
__device__ __forceinline__ void bitonic8192_desc(unsigned long long* key, int tid, int nth) {
  for (int k = 2; k <= SORT_N; k <<= 1) {
    for (int j = k >> 1; j > 0; j >>= 1) {
      for (int i = tid; i < SORT_N; i += nth) {
        int ixj = i ^ j;
        if (ixj > i) {
          unsigned long long a = key[i], b = key[ixj];
          bool up = ((i & k) == 0);   // descending segments
          if (up ? (a < b) : (a > b)) { key[i] = b; key[ixj] = a; }
        }
      }
      __syncthreads();
    }
  }
}

__global__ __launch_bounds__(512)
void k_sort_cand(const float* __restrict__ cval, const int* __restrict__ cidx,
                 const int* __restrict__ cnt, float* __restrict__ ts, int* __restrict__ ti) {
  __shared__ unsigned long long key[SORT_N];
  int bl = blockIdx.x, tid = threadIdx.x;
  int c = cnt[bl]; if (c > CAND_MAX) c = CAND_MAX;
  for (int i = tid; i < SORT_N; i += 512) {
    unsigned long long k = 0ull;
    if (i < c) {
      k = ((unsigned long long)sortkey(cval[(size_t)bl * CAND_MAX + i]) << 32)
          | (unsigned)cidx[(size_t)bl * CAND_MAX + i];
    }
    key[i] = k;
  }
  __syncthreads();
  bitonic8192_desc(key, tid, 512);
  for (int i = tid; i < PRE_NMS; i += 512) {
    float sc = -1e30f; int gi = 0;
    if (i < c) {
      unsigned long long k = key[i];
      gi = (int)(unsigned)k;
      sc = unsortkey((unsigned)(k >> 32));
    }
    ts[(size_t)bl * PRE_NMS + i] = sc;
    ti[(size_t)bl * PRE_NMS + i] = gi;
  }
}

// ---------------- box decode ----------------
struct DecPack { const float* delta[NLEV]; const float* anc[NLEV]; };

__global__ void k_decode(DecPack dp, const int* __restrict__ ti, const float* __restrict__ ts,
                         const float* __restrict__ info, float* __restrict__ tb) {
  int gid = blockIdx.x * blockDim.x + threadIdx.x;
  if (gid >= BATCH * NLEV * PRE_NMS) return;
  int b = gid / (NLEV * PRE_NMS);
  int rest = gid % (NLEV * PRE_NMS);
  int li = rest / PRE_NMS;
  float* out = tb + (size_t)gid * 4;
  if (ts[gid] < -1e29f) { out[0]=0.f; out[1]=0.f; out[2]=0.f; out[3]=0.f; return; }
  int g = ti[gid];
  int N = c_N[li];
  const float* an = dp.anc[li] + (size_t)g * 4;
  const float* d  = dp.delta[li] + ((size_t)b * N + g) * 4;
  float ah = an[2]-an[0], aw = an[3]-an[1];
  float acy = an[0] + 0.5f*ah, acx = an[1] + 0.5f*aw;
  float dh = fminf(d[2], BBOX_CLIP), dw = fminf(d[3], BBOX_CLIP);
  float cy = acy + d[0]*ah, cx = acx + d[1]*aw;
  float h = ah * expf(dh), w = aw * expf(dw);
  float hmax = info[b*5+0], wmax = info[b*5+1];
  out[0] = fminf(fmaxf(cy - 0.5f*h, 0.f), hmax);
  out[1] = fminf(fmaxf(cx - 0.5f*w, 0.f), wmax);
  out[2] = fminf(fmaxf(cy + 0.5f*h, 0.f), hmax);
  out[3] = fminf(fmaxf(cx + 0.5f*w, 0.f), wmax);
}

// ---------------- NMS (greedy, boxes in LDS) ----------------
__global__ __launch_bounds__(256)
void k_nms(const float* __restrict__ tb, float* __restrict__ ts) {
  __shared__ float bx0[PRE_NMS], bx1[PRE_NMS], bx2[PRE_NMS], bx3[PRE_NMS], ar[PRE_NMS];
  __shared__ int keep[PRE_NMS];
  int bl = blockIdx.x, tid = threadIdx.x;
  for (int i = tid; i < PRE_NMS; i += 256) {
    const float* p = tb + ((size_t)bl * PRE_NMS + i) * 4;
    float a0=p[0],a1=p[1],a2=p[2],a3=p[3];
    bx0[i]=a0; bx1[i]=a1; bx2[i]=a2; bx3[i]=a3;
    ar[i] = fmaxf(a2-a0,0.f)*fmaxf(a3-a1,0.f);
    keep[i] = 1;
  }
  __syncthreads();
  for (int i = 0; i < PRE_NMS; ++i) {
    if (keep[i]) {
      float y1=bx0[i], x1=bx1[i], y2=bx2[i], x2=bx3[i], ai=ar[i];
      for (int j = i + 1 + tid; j < PRE_NMS; j += 256) {
        if (!keep[j]) continue;
        float yy1 = fmaxf(y1, bx0[j]);
        float xx1 = fmaxf(x1, bx1[j]);
        float yy2 = fminf(y2, bx2[j]);
        float xx2 = fminf(x2, bx3[j]);
        float inter = fmaxf(yy2-yy1,0.f)*fmaxf(xx2-xx1,0.f);
        float iou = inter / (ai + ar[j] - inter + 1e-8f);
        if (iou > NMS_TH) keep[j] = 0;
      }
    }
    __syncthreads();
  }
  for (int i = tid; i < PRE_NMS; i += 256)
    if (!keep[i]) ts[(size_t)bl * PRE_NMS + i] = -1.0f;
}

// ---------------- final top-k over 5*1000 ----------------
__global__ __launch_bounds__(512)
void k_final(const float* __restrict__ ts, const float* __restrict__ tb,
             float* __restrict__ rois, float* __restrict__ rs) {
  __shared__ unsigned long long key[SORT_N];
  int b = blockIdx.x, tid = threadIdx.x;
  for (int i = tid; i < SORT_N; i += 512) {
    unsigned long long k = 0ull;
    if (i < NLEV * PRE_NMS)
      k = ((unsigned long long)sortkey(ts[(size_t)b * NLEV * PRE_NMS + i]) << 32) | (unsigned)i;
    key[i] = k;
  }
  __syncthreads();
  bitonic8192_desc(key, tid, 512);
  for (int i = tid; i < POST_NMS; i += 512) {
    unsigned long long k = key[i];
    unsigned p = (unsigned)k;
    rs[(size_t)b * POST_NMS + i] = unsortkey((unsigned)(k >> 32));
    const float* src = tb + ((size_t)b * NLEV * PRE_NMS + p) * 4;
    float* dst = rois + ((size_t)b * POST_NMS + i) * 4;
    dst[0]=src[0]; dst[1]=src[1]; dst[2]=src[2]; dst[3]=src[3];
  }
}

// ---------------- ROI align (one WG per roi, lane = channel) ----------------
struct FeatPack { const float* f[NLEV]; };

__global__ __launch_bounds__(256)
void k_roialign(FeatPack fp, const float* __restrict__ rois, unsigned short* __restrict__ Xbf) {
  __shared__ int s_li;
  __shared__ float s_box[4];
  __shared__ int s_y0[14], s_y1[14], s_x0[14], s_x1[14];
  __shared__ float s_wy[14], s_wx[14];
  int br = blockIdx.x;            // b*1000 + r
  int b = br / POST_NMS;
  int c = threadIdx.x;
  if (c == 0) {
    const float* bx = rois + (size_t)br * 4;
    float y1=bx[0], x1=bx[1], y2=bx[2], x2=bx[3];
    float area = fmaxf(y2-y1,0.f)*fmaxf(x2-x1,0.f);
    float lv = floorf(4.0f + log2f(sqrtf(area)/224.0f + 1e-8f));
    lv = fminf(fmaxf(lv, 2.0f), 6.0f);
    s_li = (int)lv - 2;
    s_box[0]=y1; s_box[1]=x1; s_box[2]=y2; s_box[3]=x2;
  }
  __syncthreads();
  int li = s_li;
  int Hl = c_H[li], Wl = c_W[li];
  if (c < 14) {
    float sc = c_stride[li];
    float t = (float)c / 13.0f;
    float ys = fminf(fmaxf((s_box[0] + (s_box[2]-s_box[0])*t)/sc, 0.f), (float)Hl - 1.f);
    int y0 = (int)floorf(ys);
    s_y0[c] = y0; s_y1[c] = min(y0+1, Hl-1); s_wy[c] = ys - (float)y0;
    float xs = fminf(fmaxf((s_box[1] + (s_box[3]-s_box[1])*t)/sc, 0.f), (float)Wl - 1.f);
    int x0 = (int)floorf(xs);
    s_x0[c] = x0; s_x1[c] = min(x0+1, Wl-1); s_wx[c] = xs - (float)x0;
  }
  __syncthreads();
  const float* fb = fp.f[li] + (size_t)b * Hl * Wl * CCH;
  unsigned short* xout = Xbf + (size_t)br * DIMX;
  for (int oy = 0; oy < 7; ++oy) {
    for (int ox = 0; ox < 7; ++ox) {
      float acc = 0.f;
      #pragma unroll
      for (int dy = 0; dy < 2; ++dy) {
        int sy = 2*oy+dy;
        int ya = s_y0[sy], yb = s_y1[sy];
        float fy = s_wy[sy];
        #pragma unroll
        for (int dx = 0; dx < 2; ++dx) {
          int sx = 2*ox+dx;
          int xa = s_x0[sx], xb = s_x1[sx];
          float fx = s_wx[sx];
          float v00 = fb[((size_t)ya*Wl+xa)*CCH + c];
          float v01 = fb[((size_t)ya*Wl+xb)*CCH + c];
          float v10 = fb[((size_t)yb*Wl+xa)*CCH + c];
          float v11 = fb[((size_t)yb*Wl+xb)*CCH + c];
          acc += v00*(1.f-fy)*(1.f-fx) + v01*(1.f-fy)*fx + v10*fy*(1.f-fx) + v11*fy*fx;
        }
      }
      xout[(oy*7+ox)*CCH + c] = f2bf(acc * 0.25f);
    }
  }
}

// ---------------- f32 [K][N] -> bf16 [N][K] tiled transpose ----------------
__global__ __launch_bounds__(256)
void k_cvt_transpose(const float* __restrict__ in, unsigned short* __restrict__ outT,
                     int K, int N) {
  __shared__ unsigned short tile[32][33];
  int n0 = blockIdx.x * 32, k0 = blockIdx.y * 32;
  int tx = threadIdx.x, ty = threadIdx.y;        // 32 x 8
  #pragma unroll
  for (int i = 0; i < 4; ++i) {
    int k = k0 + ty + i * 8;
    tile[tx][ty + i * 8] = f2bf(in[(size_t)k * N + n0 + tx]);
  }
  __syncthreads();
  #pragma unroll
  for (int i = 0; i < 4; ++i) {
    int n = n0 + ty + i * 8;
    outT[(size_t)n * K + k0 + tx] = tile[ty + i * 8][tx];
  }
}

// ---------------- WMMA bf16 GEMM: C = relu(A*BT^T + bias), bf16 out ----------
// 64x128 tile / WG, 256 threads = 8 wave32, each wave: 16x64 strip -> 4 wmma
// per 32-K step (32 wmma / WG / step); double-buffered LDS; uint4 tile loads;
// low register pressure (4 accs + 3 staging uint4) so nothing spills.
#define BM 64
#define BN 128
#define BK 32

__global__ __launch_bounds__(256)
void k_gemm_bf16(const unsigned short* __restrict__ A,   // MROWS x K (bf16, padded)
                 const unsigned short* __restrict__ BT,  // N x K (bf16, transposed)
                 const float* __restrict__ bias,
                 unsigned short* __restrict__ Cout,      // MROWS x N (bf16)
                 int N, int K) {
  __shared__ unsigned short sA[2][BM * BK];   // [m][k]
  __shared__ unsigned short sB[2][BN * BK];   // [n][k]
  int tid = threadIdx.x;
  int wave = tid >> 5, lane = tid & 31;
  int wy = wave & 3, wx = wave >> 2;          // wave -> (m strip, n half)
  int bm = blockIdx.x * BM, bn = blockIdx.y * BN;
  int h = lane >> 4, l15 = lane & 15;
  int ml = wy * 16 + l15;

  v8f acc[4];
  #pragma unroll
  for (int j = 0; j < 4; ++j)
    #pragma unroll
    for (int e = 0; e < 8; ++e) acc[j][e] = 0.f;

  const int T = K / BK;

  // per-thread tile-load addresses (1 uint4 of A, 2 uint4 of B per K-step)
  int ar = tid >> 2, ac4 = tid & 3;
  const unsigned short* Aptr = A + (size_t)(bm + ar) * K + ac4 * 8;
  int bi0 = tid * 2, bi1 = tid * 2 + 1;
  const unsigned short* Bptr0 = BT + (size_t)(bn + (bi0 >> 2)) * K + (bi0 & 3) * 8;
  const unsigned short* Bptr1 = BT + (size_t)(bn + (bi1 >> 2)) * K + (bi1 & 3) * 8;

  // preload tile 0
  ((uint4*)sA[0])[tid]     = *(const uint4*)Aptr;
  ((uint4*)sB[0])[bi0]     = *(const uint4*)Bptr0;
  ((uint4*)sB[0])[bi1]     = *(const uint4*)Bptr1;
  __syncthreads();

  for (int it = 0; it < T; ++it) {
    int cur = it & 1, nxt = cur ^ 1;
    int kn = (it + 1) * BK;
    bool hn = (it + 1) < T;
    uint4 a_st, b_st0, b_st1;
    if (hn) {
      a_st  = *(const uint4*)(Aptr  + kn);
      b_st0 = *(const uint4*)(Bptr0 + kn);
      b_st1 = *(const uint4*)(Bptr1 + kn);
    }

    union U { v16bf v; unsigned int u[8]; };
    // A fragment (16x32 bf16, documented lane/VGPR layout)
    U af;
    const unsigned int* rowA = (const unsigned int*)(sA[cur]) + ml * 16;
    #pragma unroll
    for (int v = 0; v < 8; ++v) {
      int idx = (v < 4) ? (v + 4 * h) : (v + 4 + 4 * h);
      af.u[v] = rowA[idx];
    }
    // software-pipelined B fragments: load j+1 between wmma(j)
    const unsigned int* sBu = (const unsigned int*)(sB[cur]);
    U bfr[2];
    {
      const unsigned int* rowB = sBu + (wx * 64 + l15) * 16;
      #pragma unroll
      for (int v = 0; v < 8; ++v) bfr[0].u[v] = rowB[(v < 4) ? (v + 4 * h) : (v + 4 + 4 * h)];
    }
    #pragma unroll
    for (int j = 0; j < 4; ++j) {
      if (j < 3) {
        const unsigned int* rowB = sBu + (wx * 64 + (j + 1) * 16 + l15) * 16;
        #pragma unroll
        for (int v = 0; v < 8; ++v) bfr[(j + 1) & 1].u[v] = rowB[(v < 4) ? (v + 4 * h) : (v + 4 + 4 * h)];
      }
      acc[j] = __builtin_amdgcn_wmma_f32_16x16x32_bf16(
          false, af.v, false, bfr[j & 1].v, (short)0, acc[j], false, false);
    }

    if (hn) {
      ((uint4*)sA[nxt])[tid] = a_st;
      ((uint4*)sB[nxt])[bi0] = b_st0;
      ((uint4*)sB[nxt])[bi1] = b_st1;
    }
    __syncthreads();
  }

  // epilogue: +bias, relu, bf16 store (C layout: VGPR r -> M = r + 8*h)
  #pragma unroll
  for (int j = 0; j < 4; ++j) {
    int n = bn + wx * 64 + j * 16 + l15;
    float bv = bias[n];
    #pragma unroll
    for (int r = 0; r < 8; ++r) {
      int m = bm + wy * 16 + r + h * 8;
      float x = acc[j][r] + bv;
      x = x > 0.f ? x : 0.f;
      Cout[(size_t)m * N + n] = f2bf(x);
    }
  }
}

// ---------------- heads: cls (91) + box (364) ----------------
__global__ void k_heads(const unsigned short* __restrict__ h2,
                        const float* __restrict__ wc, const float* __restrict__ bc,
                        const float* __restrict__ wb, const float* __restrict__ bb,
                        float* __restrict__ out_cls, float* __restrict__ out_box) {
  int gid = blockIdx.x * blockDim.x + threadIdx.x;
  if (gid >= MVALID * (NCLS + 4 * NCLS)) return;
  int m = gid / (NCLS + 4 * NCLS);
  int j = gid % (NCLS + 4 * NCLS);
  const unsigned short* hr = h2 + (size_t)m * MLP;
  float acc = 0.f;
  if (j < NCLS) {
    for (int k = 0; k < MLP; ++k) acc += bf2f(hr[k]) * wc[(size_t)k * NCLS + j];
    out_cls[(size_t)m * NCLS + j] = acc + bc[j];
  } else {
    int jj = j - NCLS;
    for (int k = 0; k < MLP; ++k) acc += bf2f(hr[k]) * wb[(size_t)k * 4 * NCLS + jj];
    out_box[(size_t)m * 4 * NCLS + jj] = acc + bb[jj];
  }
}

// ---------------- host launch ----------------
extern "C" void kernel_launch(void* const* d_in, const int* in_sizes, int n_in,
                              void* d_out, int out_size, void* d_ws, size_t ws_size,
                              hipStream_t stream) {
  (void)in_sizes; (void)n_in; (void)out_size; (void)ws_size;
  static const int hN[NLEV]   = {209664,52416,13104,3276,819};
  static const int hOff[NLEV] = {0,209664,262080,275184,278460};

  const float *feat[NLEV], *score[NLEV], *delta[NLEV], *anc[NLEV];
  for (int l = 0; l < NLEV; ++l) {
    feat[l]  = (const float*)d_in[4*l + 0];
    score[l] = (const float*)d_in[4*l + 1];
    delta[l] = (const float*)d_in[4*l + 2];
    anc[l]   = (const float*)d_in[4*l + 3];
  }
  const float* info = (const float*)d_in[20];
  const float* w1 = (const float*)d_in[21]; const float* b1 = (const float*)d_in[22];
  const float* w2 = (const float*)d_in[23]; const float* b2 = (const float*)d_in[24];
  const float* wc = (const float*)d_in[25]; const float* bc = (const float*)d_in[26];
  const float* wb = (const float*)d_in[27]; const float* bb = (const float*)d_in[28];

  float* out = (float*)d_out;
  float* out_cls  = out;                       // 2*1000*91   = 182000
  float* out_box  = out + 182000;              // 2*1000*364  = 728000
  float* out_rois = out + 910000;              // 2*1000*4    = 8000
  float* out_rs   = out + 918000;              // 2*1000      = 2000

  char* p = (char*)d_ws;
  auto alloc = [&](size_t bytes) -> void* {
    void* r = (void*)p;
    p += (bytes + 255) & ~(size_t)255;
    return r;
  };
  float* s_flat = (float*)alloc((size_t)BATCH * TOTA * 4);
  int*   hist   = (int*)alloc((size_t)BATCH * NLEV * 1024 * 4);
  int*   thr    = (int*)alloc(BATCH * NLEV * 4);
  int*   ccnt   = (int*)alloc(BATCH * NLEV * 4);
  float* cval   = (float*)alloc((size_t)BATCH * NLEV * CAND_MAX * 4);
  int*   cidx   = (int*)alloc((size_t)BATCH * NLEV * CAND_MAX * 4);
  float* ts     = (float*)alloc((size_t)BATCH * NLEV * PRE_NMS * 4);
  int*   ti     = (int*)alloc((size_t)BATCH * NLEV * PRE_NMS * 4);
  float* tb     = (float*)alloc((size_t)BATCH * NLEV * PRE_NMS * 16);
  unsigned short* Xbf  = (unsigned short*)alloc((size_t)MROWS * DIMX * 2);
  unsigned short* W1T  = (unsigned short*)alloc((size_t)DIMX * MLP * 2);  // [N][K]
  unsigned short* W2T  = (unsigned short*)alloc((size_t)MLP * MLP * 2);   // [N][K]
  unsigned short* h1   = (unsigned short*)alloc((size_t)MROWS * MLP * 2);
  unsigned short* h2   = (unsigned short*)alloc((size_t)MROWS * MLP * 2);

  hipMemsetAsync(hist, 0, (size_t)BATCH * NLEV * 1024 * 4, stream);
  hipMemsetAsync(ccnt, 0, BATCH * NLEV * 4, stream);
  // zero the padded activation rows so the GEMM needs no bounds checks
  hipMemsetAsync(Xbf + (size_t)MVALID * DIMX, 0,
                 (size_t)(MROWS - MVALID) * DIMX * 2, stream);

  for (int l = 0; l < NLEV; ++l) {
    int n = BATCH * hN[l];
    int g = (n + 255) / 256;
    k_sigmoid<<<g, 256, 0, stream>>>(score[l], s_flat, hOff[l], hN[l]);
  }
  for (int l = 0; l < NLEV; ++l) {
    int n = BATCH * hN[l];
    int g = (n + 255) / 256;
    k_hist<<<g, 256, 0, stream>>>(s_flat, hist, hOff[l], hN[l], l);
  }
  k_thresh<<<BATCH * NLEV, 32, 0, stream>>>(hist, thr);
  for (int l = 0; l < NLEV; ++l) {
    int n = BATCH * hN[l];
    int g = (n + 255) / 256;
    k_compact<<<g, 256, 0, stream>>>(s_flat, thr, ccnt, cval, cidx, hOff[l], hN[l], l);
  }
  k_sort_cand<<<BATCH * NLEV, 512, 0, stream>>>(cval, cidx, ccnt, ts, ti);

  DecPack dp;
  for (int l = 0; l < NLEV; ++l) { dp.delta[l] = delta[l]; dp.anc[l] = anc[l]; }
  {
    int n = BATCH * NLEV * PRE_NMS;
    k_decode<<<(n + 255) / 256, 256, 0, stream>>>(dp, ti, ts, info, tb);
  }
  k_nms<<<BATCH * NLEV, 256, 0, stream>>>(tb, ts);
  k_final<<<BATCH, 512, 0, stream>>>(ts, tb, out_rois, out_rs);

  FeatPack fp;
  for (int l = 0; l < NLEV; ++l) fp.f[l] = feat[l];
  k_roialign<<<BATCH * POST_NMS, 256, 0, stream>>>(fp, out_rois, Xbf);

  // transpose-convert weights once per launch (L2-resident thereafter)
  {
    dim3 b(32, 8);
    dim3 gw1(MLP / 32, DIMX / 32);
    k_cvt_transpose<<<gw1, b, 0, stream>>>(w1, W1T, DIMX, MLP);
    dim3 gw2(MLP / 32, MLP / 32);
    k_cvt_transpose<<<gw2, b, 0, stream>>>(w2, W2T, MLP, MLP);
  }

  dim3 g1(MROWS / BM, MLP / BN);
  k_gemm_bf16<<<g1, 256, 0, stream>>>(Xbf, W1T, b1, h1, MLP, DIMX);
  k_gemm_bf16<<<g1, 256, 0, stream>>>(h1, W2T, b2, h2, MLP, MLP);

  {
    int n = MVALID * (NCLS + 4 * NCLS);
    k_heads<<<(n + 255) / 256, 256, 0, stream>>>(h2, wc, bc, wb, bb, out_cls, out_box);
  }
}
